// ODEGRU_1219770712315
// MI455X (gfx1250) — matrix operand order repaired and checked by reading, not compiled, not run on hardware
//
#include <hip/hip_runtime.h>
#include <hip/hip_bf16.h>

// Problem sizes (from reference)
#define SEQ    256
#define BATCH  512
#define DIN    54
#define DPAD   64      // D padded to 64 for K-alignment (zeros)
#define HDIM   256
#define DHID   64
#define NODE   8
#define MTILE  16      // batch rows per workgroup
#define NTHR   128     // 4 waves (wave32)

typedef __bf16 bf16_t;
typedef __attribute__((ext_vector_type(16))) __bf16 v16bf;
typedef __attribute__((ext_vector_type(8)))  __bf16 v8bf;
typedef __attribute__((ext_vector_type(8)))  float  v8f;

// ---- WMMA fragment builders (wave32, 16x16x32 bf16) ----
// A (16x32, MxK): lane = row (lane&15), kh = lane>>4.
//   elements 0..7  -> K = koff + kh*8 + e
//   elements 8..15 -> K = koff + 16 + kh*8 + (e-8)
__device__ __forceinline__ v16bf fragA_bf16(const bf16_t* p0, int row, int ld, int koff, int kh) {
  const bf16_t* p = p0 + row * ld + koff + kh * 8;
  v8bf lo = *(const v8bf*)(p);
  v8bf hi = *(const v8bf*)(p + 16);
  return __builtin_shufflevector(lo, hi, 0,1,2,3,4,5,6,7,8,9,10,11,12,13,14,15);
}

// B (32x16, KxN), source stored as src[n*ld + k] (W row-major, B = W^T):
// lane = column (lane&15), kh = lane>>4; same K order as A:
//   elements 0..7 -> K = koff + kh*8 + e, elements 8..15 -> K = koff + 16 + kh*8 + (e-8)
__device__ __forceinline__ v16bf fragB_bf16(const bf16_t* p0, int col, int ld, int koff, int kh) {
  const bf16_t* p = p0 + col * ld + koff + kh * 8;
  v8bf lo = *(const v8bf*)(p);
  v8bf hi = *(const v8bf*)(p + 16);
  return __builtin_shufflevector(lo, hi, 0,1,2,3,4,5,6,7,8,9,10,11,12,13,14,15);
}

__device__ __forceinline__ v8f wmma_bf16(v16bf a, v16bf b, v8f c) {
  return __builtin_amdgcn_wmma_f32_16x16x32_bf16(false, a, false, b, (short)0, c, false, false);
}

__device__ __forceinline__ float sigmoidf_(float v) { return 1.0f / (1.0f + __expf(-v)); }

// ---- prep: convert weights to bf16 in workspace (W_ih K-padded 54->64) ----
__global__ void odegru_convert_weights(const float* __restrict__ Wih,
                                       const float* __restrict__ Whh,
                                       bf16_t* __restrict__ gWih,
                                       bf16_t* __restrict__ gWhh) {
  const int stride = gridDim.x * blockDim.x;
  const int tid = blockIdx.x * blockDim.x + threadIdx.x;
  for (int i = tid; i < 3 * HDIM * DPAD; i += stride) {
    const int n = i >> 6, k = i & 63;
    gWih[i] = (bf16_t)((k < DIN) ? Wih[n * DIN + k] : 0.0f);
  }
  for (int i = tid; i < 3 * HDIM * HDIM; i += stride) {
    gWhh[i] = (bf16_t)Whh[i];
  }
}

// ---- persistent kernel: each block owns MTILE batch rows for the full sequence ----
__global__ __launch_bounds__(NTHR, 1) void odegru_kernel(
    const float* __restrict__ x, const float* __restrict__ t,
    const bf16_t* __restrict__ gWih, const bf16_t* __restrict__ gWhh,
    const float* __restrict__ b_ih, const float* __restrict__ b_hh,
    const float* __restrict__ W1f, const float* __restrict__ b1,
    const float* __restrict__ W2f, const float* __restrict__ b2,
    float* __restrict__ out) {
  // Dynamic LDS carve-out (~204 KB; CDNA5 allows up to 320 KB per workgroup)
  extern __shared__ __align__(32) char smem[];
  float*  sK  = (float*)smem;                 // [6][MTILE*HDIM] f32  (k1..k6)
  float*  sHa = sK + 6 * MTILE * HDIM;        // [MTILE*HDIM]    f32  (h ping)
  float*  sHb = sHa + MTILE * HDIM;           // [MTILE*HDIM]    f32  (h pong)
  bf16_t* sW1 = (bf16_t*)(sHb + MTILE * HDIM);// [DHID*HDIM]     bf16 (B = W1^T source)
  bf16_t* sW2 = sW1 + DHID * HDIM;            // [HDIM*DHID]     bf16 (B = W2^T source)
  bf16_t* sHe = sW2 + HDIM * DHID;            // [MTILE*HDIM]    bf16 (A-operand staging)
  bf16_t* sZ  = sHe + MTILE * HDIM;           // [MTILE*DHID]    bf16 (tanh layer out)
  bf16_t* sX  = sZ + MTILE * DHID;            // [MTILE*DPAD]    bf16 (staged x tile)

  const int tid  = threadIdx.x;
  const int lane = tid & 31;
  const int wave = tid >> 5;
  const int ln   = lane & 15;   // column within a 16-wide tile
  const int kh   = lane >> 4;   // K-half selector
  const int row0 = blockIdx.x * MTILE;

  // Load drift weights once (reused 48x per sequence step)
  for (int i = tid; i < DHID * HDIM; i += NTHR) sW1[i] = (bf16_t)W1f[i];
  for (int i = tid; i < HDIM * DHID; i += NTHR) sW2[i] = (bf16_t)W2f[i];
  for (int i = tid; i < MTILE * HDIM; i += NTHR) sHa[i] = 0.0f;  // h0 = 0
  __syncthreads();

  // dopri5 tableau
  const float AT[5][5] = {
    {0.2f, 0.f, 0.f, 0.f, 0.f},
    {3.f/40.f, 9.f/40.f, 0.f, 0.f, 0.f},
    {44.f/45.f, -56.f/15.f, 32.f/9.f, 0.f, 0.f},
    {19372.f/6561.f, -25360.f/2187.f, 64448.f/6561.f, -212.f/729.f, 0.f},
    {9017.f/3168.f, -355.f/33.f, 46732.f/5247.f, 49.f/176.f, -5103.f/18656.f}};
  const float BT[6] = {35.f/384.f, 0.f, 500.f/1113.f, 125.f/192.f, -2187.f/6784.f, 11.f/84.f};

  for (int step = 0; step < SEQ; ++step) {
    float* hprev = (step & 1) ? sHb : sHa;   // state entering this step
    float* h     = (step & 1) ? sHa : sHb;   // state produced by this step

    const int si = SEQ - 1 - step;           // reversed-time index
    const float t0 = t[si];
    const float t1 = (step < SEQ - 1) ? t[si - 1] : t[si];
    const float dt = (t1 - t0) / (float)NODE;

    // stage x tile (zero-pad K 54->64) and bf16 mirror of hprev
    for (int idx = tid; idx < MTILE * DPAD; idx += NTHR) {
      const int m = idx >> 6, d = idx & 63;
      const float v = (d < DIN) ? x[(long)si * BATCH * DIN + (long)(row0 + m) * DIN + d] : 0.0f;
      sX[idx] = (bf16_t)v;
    }
    for (int idx = tid; idx < MTILE * HDIM; idx += NTHR) sHe[idx] = (bf16_t)hprev[idx];
    __syncthreads();

    // ---------------- GRU cell ----------------
    // r/z gates: x-GEMM and h-GEMM accumulate into the SAME WMMA accumulator
    // (sigmoid(i+h)); only the n gate needs i_n and h_n kept apart.
#pragma unroll 1
    for (int tc = 0; tc < 4; ++tc) {
      const int ncol = (wave + 4 * tc) * 16 + ln;  // 0..255
      v8f aR  = {0.f,0.f,0.f,0.f,0.f,0.f,0.f,0.f};
      v8f aZ  = aR, aIN = aR, aHN = aR;
#pragma unroll
      for (int kk = 0; kk < DPAD; kk += 32) {
        const v16bf a = fragA_bf16(sX, ln, DPAD, kk, kh);
        aR  = wmma_bf16(a, fragB_bf16(gWih, ncol,            DPAD, kk, kh), aR);
        aZ  = wmma_bf16(a, fragB_bf16(gWih, HDIM + ncol,     DPAD, kk, kh), aZ);
        aIN = wmma_bf16(a, fragB_bf16(gWih, 2 * HDIM + ncol, DPAD, kk, kh), aIN);
      }
#pragma unroll 1
      for (int kk = 0; kk < HDIM; kk += 32) {
        const v16bf a = fragA_bf16(sHe, ln, HDIM, kk, kh);
        aR  = wmma_bf16(a, fragB_bf16(gWhh, ncol,            HDIM, kk, kh), aR);
        aZ  = wmma_bf16(a, fragB_bf16(gWhh, HDIM + ncol,     HDIM, kk, kh), aZ);
        aHN = wmma_bf16(a, fragB_bf16(gWhh, 2 * HDIM + ncol, HDIM, kk, kh), aHN);
      }
      const float brz = b_ih[ncol] + b_hh[ncol];
      const float bzz = b_ih[HDIM + ncol] + b_hh[HDIM + ncol];
      const float bin = b_ih[2 * HDIM + ncol];
      const float bhn = b_hh[2 * HDIM + ncol];
#pragma unroll
      for (int r = 0; r < 8; ++r) {
        const int m = r + 8 * kh;
        const float hold = hprev[m * HDIM + ncol];
        const float rg = sigmoidf_(aR[r] + brz);
        const float zg = sigmoidf_(aZ[r] + bzz);
        const float ng = tanhf(aIN[r] + bin + rg * (aHN[r] + bhn));
        h[m * HDIM + ncol] = (1.0f - zg) * ng + zg * hold;  // write to pong buffer
      }
    }
    __syncthreads();

    // ---------------- 8 dopri5 steps (in-place on h) ----------------
    for (int ode = 0; ode < NODE; ++ode) {
#pragma unroll
      for (int j = 0; j < 6; ++j) {
        // heval = h + dt * sum_l AT[j-1][l] * k_l   (bf16, ready as A operand)
        for (int idx = tid; idx < MTILE * HDIM; idx += NTHR) {
          float v = h[idx];
#pragma unroll
          for (int l = 0; l < 5; ++l)
            if (l < j) v += (dt * AT[j - 1][l]) * sK[l * MTILE * HDIM + idx];
          sHe[idx] = (bf16_t)v;
        }
        __syncthreads();
        // z = tanh(heval @ W1^T + b1): 16x64 out, one 16x16 tile per wave, K=256
        {
          const int ncol = wave * 16 + ln;  // 0..63
          v8f acc = {0.f,0.f,0.f,0.f,0.f,0.f,0.f,0.f};
#pragma unroll
          for (int kk = 0; kk < HDIM; kk += 32)
            acc = wmma_bf16(fragA_bf16(sHe, ln, HDIM, kk, kh),
                            fragB_bf16(sW1, ncol, HDIM, kk, kh), acc);
          const float bb = b1[ncol];
#pragma unroll
          for (int r = 0; r < 8; ++r)
            sZ[(r + 8 * kh) * DHID + ncol] = (bf16_t)tanhf(acc[r] + bb);
        }
        __syncthreads();
        // k_j = z @ W2^T + b2: 16x256 out, 4 tiles per wave, K=64
#pragma unroll 1
        for (int tc = 0; tc < 4; ++tc) {
          const int ncol = (wave + 4 * tc) * 16 + ln;  // 0..255
          v8f acc = {0.f,0.f,0.f,0.f,0.f,0.f,0.f,0.f};
#pragma unroll
          for (int kk = 0; kk < DHID; kk += 32)
            acc = wmma_bf16(fragA_bf16(sZ, ln, DHID, kk, kh),
                            fragB_bf16(sW2, ncol, DHID, kk, kh), acc);
          const float bb = b2[ncol];
#pragma unroll
          for (int r = 0; r < 8; ++r)
            sK[j * MTILE * HDIM + (r + 8 * kh) * HDIM + ncol] = acc[r] + bb;
        }
        __syncthreads();
      }
      // h += dt * (b . k)
      for (int idx = tid; idx < MTILE * HDIM; idx += NTHR) {
        float v = h[idx];
#pragma unroll
        for (int l = 0; l < 6; ++l) v += (dt * BT[l]) * sK[l * MTILE * HDIM + idx];
        h[idx] = v;
      }
      __syncthreads();
    }

    // hs[step] -> out row si (output is hs[::-1])
    for (int idx = tid; idx < MTILE * HDIM; idx += NTHR) {
      const int m = idx >> 8, cc = idx & (HDIM - 1);
      out[(long)si * BATCH * HDIM + (long)(row0 + m) * HDIM + cc] = h[idx];
    }
    __syncthreads();
  }
}

extern "C" void kernel_launch(void* const* d_in, const int* in_sizes, int n_in,
                              void* d_out, int out_size, void* d_ws, size_t ws_size,
                              hipStream_t stream) {
  const float* x    = (const float*)d_in[0];
  const float* t    = (const float*)d_in[1];
  const float* Wih  = (const float*)d_in[2];
  const float* Whh  = (const float*)d_in[3];
  const float* b_ih = (const float*)d_in[4];
  const float* b_hh = (const float*)d_in[5];
  const float* W1   = (const float*)d_in[6];
  const float* b1   = (const float*)d_in[7];
  const float* W2   = (const float*)d_in[8];
  const float* b2   = (const float*)d_in[9];

  bf16_t* gWih = (bf16_t*)d_ws;                    // [768][64] bf16, K-padded
  bf16_t* gWhh = gWih + 3 * HDIM * DPAD;           // [768][256] bf16 (L2-resident)

  odegru_convert_weights<<<128, 256, 0, stream>>>(Wih, Whh, gWih, gWhh);

  const size_t smem =
      (size_t)(6 * MTILE * HDIM + 2 * MTILE * HDIM) * sizeof(float) +
      (size_t)(DHID * HDIM + HDIM * DHID + MTILE * HDIM + MTILE * DHID + MTILE * DPAD) * sizeof(bf16_t);

  odegru_kernel<<<BATCH / MTILE, NTHR, smem, stream>>>(
      x, t, gWih, gWhh, b_ih, b_hh, W1, b1, W2, b2, (float*)d_out);
}
